// GraphSAGE_36464272343145
// MI455X (gfx1250) — compile-verified, matrix operand order checked by reading, not run on hardware
//
#include <hip/hip_runtime.h>

#define IN_FEATS 128
#define HIDDEN   128
#define NCLS     64

typedef __attribute__((ext_vector_type(16))) __bf16 v16bf;
typedef __attribute__((ext_vector_type(8)))  __bf16 v8bf;
typedef __attribute__((ext_vector_type(8)))  float  v8f;

// Fragment layout per ISA 7.12.2 (16-bit, 16x32): for each lane,
// elements 0..7 -> K = base+0..7, elements 8..15 -> K = base+16..23,
// where base = (lane>>4)*8. `p` already points at K = kb + base.

// f32 source (node data: unique per wave, convert on the fly)
__device__ __forceinline__ v16bf load_frag_f32(const float* __restrict__ p,
                                               float scale) {
  v16bf a;
#pragma unroll
  for (int i = 0; i < 8; ++i) a[i] = (__bf16)(p[i] * scale);
#pragma unroll
  for (int i = 0; i < 8; ++i) a[8 + i] = (__bf16)(p[16 + i] * scale);
  return a;
}

// bf16 source (weights: pre-converted once, two b128 loads, zero VALU)
__device__ __forceinline__ v16bf load_frag_bf16(const __bf16* __restrict__ p) {
  v8bf lo = *(const v8bf*)(p);        // K = base + 0..7
  v8bf hi = *(const v8bf*)(p + 16);   // K = base + 16..23
  v16bf a;
#pragma unroll
  for (int i = 0; i < 8; ++i) { a[i] = lo[i]; a[8 + i] = hi[i]; }
  return a;
}

// --- tiny pre-pass: stage weights as bf16 ------------------------------------

__global__ void __launch_bounds__(256)
cvt_bf16_kernel(const float* __restrict__ in, __bf16* __restrict__ out, int n) {
  int i = blockIdx.x * blockDim.x + threadIdx.x;
  if (i < n) out[i] = (__bf16)in[i];
}

// --- edge kernels ------------------------------------------------------------

__global__ void __launch_bounds__(256)
degree_kernel(const int* __restrict__ dst, float* __restrict__ deg, int E) {
  int e = blockIdx.x * blockDim.x + threadIdx.x;
  if (e < E) atomicAdd(&deg[dst[e]], 1.0f);
}

// one wave per edge: lane L handles features [4L, 4L+4)
__global__ void __launch_bounds__(256)
scatter_feat_kernel(const float* __restrict__ feat, const int* __restrict__ src,
                    const int* __restrict__ dst, float* __restrict__ agg, int E) {
  long long t = (long long)blockIdx.x * blockDim.x + threadIdx.x;
  int e = (int)(t >> 5);
  if (e >= E) return;
  int lane = threadIdx.x & 31;
  int s = src[e], d = dst[e];
  const float4 v = *(const float4*)(feat + (size_t)s * IN_FEATS + lane * 4);
  float* o = agg + (size_t)d * IN_FEATS + lane * 4;
  atomicAdd(o + 0, v.x);
  atomicAdd(o + 1, v.y);
  atomicAdd(o + 2, v.z);
  atomicAdd(o + 3, v.w);
}

// one wave per edge: lane L handles classes [2L, 2L+2)
__global__ void __launch_bounds__(256)
scatter_out_kernel(const float* __restrict__ hout, const int* __restrict__ src,
                   const int* __restrict__ dst, float* __restrict__ out, int E) {
  long long t = (long long)blockIdx.x * blockDim.x + threadIdx.x;
  int e = (int)(t >> 5);
  if (e >= E) return;
  int lane = threadIdx.x & 31;
  int s = src[e], d = dst[e];
  const float2 v = *(const float2*)(hout + (size_t)d * NCLS + lane * 2);
  float* o = out + (size_t)s * NCLS + lane * 2;
  atomicAdd(o + 0, v.x);
  atomicAdd(o + 1, v.y);
}

// --- dense kernels (WMMA) ----------------------------------------------------

// h = (relu(agg/deg * Wl^T + feat * Wr^T)) * mask1 * mask2
// One wave computes one 16x16 output tile; K=128 via 4 x (16x16x32 bf16) per operand.
__global__ void __launch_bounds__(256)
gemm1_kernel(const float* __restrict__ agg, const float* __restrict__ feat,
             const float* __restrict__ deg, const __bf16* __restrict__ Wl,
             const __bf16* __restrict__ Wr, const float* __restrict__ mask1,
             const float* __restrict__ mask2, float* __restrict__ h, int nNodes) {
  const int lane   = threadIdx.x & 31;
  const int waveId = blockIdx.x * (blockDim.x >> 5) + (threadIdx.x >> 5);
  const int tn = waveId & 7;          // HIDDEN/16 = 8 column tiles
  const int tm = waveId >> 3;
  const int m0 = tm << 4;
  if (m0 >= nNodes) return;
  const int n0   = tn << 4;
  const int sub  = lane >> 4;         // 0 or 1: selects K sub-block
  int rowA = m0 + (lane & 15);
  if (rowA >= nNodes) rowA = nNodes - 1;   // partial-tile clamp (reads only)
  const int colB = n0 + (lane & 15);
  const float rdeg = 1.0f / fmaxf(deg[rowA], 1.0f);   // fused mean-normalization

  v8f c = {};
#pragma unroll
  for (int kb = 0; kb < IN_FEATS; kb += 32) {
    const float*  pa = agg  + (size_t)rowA * IN_FEATS + kb + sub * 8;
    const float*  pf = feat + (size_t)rowA * IN_FEATS + kb + sub * 8;
    const __bf16* pl = Wl   + (size_t)colB * IN_FEATS + kb + sub * 8;
    const __bf16* pr = Wr   + (size_t)colB * IN_FEATS + kb + sub * 8;
    v16bf aA = load_frag_f32(pa, rdeg);
    v16bf bL = load_frag_bf16(pl);
    c = __builtin_amdgcn_wmma_f32_16x16x32_bf16(false, aA, false, bL,
                                                (short)0, c, false, false);
    v16bf aF = load_frag_f32(pf, 1.0f);
    v16bf bR = load_frag_bf16(pr);
    c = __builtin_amdgcn_wmma_f32_16x16x32_bf16(false, aF, false, bR,
                                                (short)0, c, false, false);
  }
  // C/D f32 layout: lane (0..15 | 16..31) = col, VGPR r = row (m0 + sub*8 + r)
  const int col = n0 + (lane & 15);
  const int rb  = m0 + sub * 8;
#pragma unroll
  for (int r = 0; r < 8; ++r) {
    int row = rb + r;
    if (row < nNodes) {
      size_t idx = (size_t)row * HIDDEN + col;
      h[idx] = fmaxf(c[r], 0.0f) * mask1[idx] * mask2[idx];
    }
  }
}

// hout = h * Wout^T   ([N,128] x [128,64])
__global__ void __launch_bounds__(256)
gemm2_kernel(const float* __restrict__ h, const __bf16* __restrict__ Wout,
             float* __restrict__ hout, int nNodes) {
  const int lane   = threadIdx.x & 31;
  const int waveId = blockIdx.x * (blockDim.x >> 5) + (threadIdx.x >> 5);
  const int tn = waveId & 3;          // NCLS/16 = 4 column tiles
  const int tm = waveId >> 2;
  const int m0 = tm << 4;
  if (m0 >= nNodes) return;
  const int n0  = tn << 4;
  const int sub = lane >> 4;
  int rowA = m0 + (lane & 15);
  if (rowA >= nNodes) rowA = nNodes - 1;
  const int colB = n0 + (lane & 15);

  v8f c = {};
#pragma unroll
  for (int kb = 0; kb < HIDDEN; kb += 32) {
    const float*  pa = h    + (size_t)rowA * HIDDEN + kb + sub * 8;
    const __bf16* pw = Wout + (size_t)colB * HIDDEN + kb + sub * 8;
    v16bf aH = load_frag_f32(pa, 1.0f);
    v16bf bW = load_frag_bf16(pw);
    c = __builtin_amdgcn_wmma_f32_16x16x32_bf16(false, aH, false, bW,
                                                (short)0, c, false, false);
  }
  const int col = n0 + (lane & 15);
  const int rb  = m0 + sub * 8;
#pragma unroll
  for (int r = 0; r < 8; ++r) {
    int row = rb + r;
    if (row < nNodes) hout[(size_t)row * NCLS + col] = c[r];
  }
}

// --- host launcher -----------------------------------------------------------

extern "C" void kernel_launch(void* const* d_in, const int* in_sizes, int n_in,
                              void* d_out, int out_size, void* d_ws, size_t ws_size,
                              hipStream_t stream) {
  const float* feat  = (const float*)d_in[0];
  const int*   ei    = (const int*)d_in[1];
  const float* Wl    = (const float*)d_in[2];
  const float* Wr    = (const float*)d_in[3];
  const float* Wout  = (const float*)d_in[4];
  const float* mask1 = (const float*)d_in[5];
  const float* mask2 = (const float*)d_in[6];
  float* out = (float*)d_out;

  const int N = in_sizes[0] / IN_FEATS;     // 100000
  const int E = in_sizes[1] / 2;            // 3200000
  const int* src = ei;
  const int* dst = ei + E;

  // workspace: deg[N] | agg[N,128] | h[N,128] | Wl_bf | Wr_bf | Wout_bf
  // hout[N,64] reuses agg storage (stream-ordered: gemm1 consumed agg already)
  char* ws = (char*)d_ws;
  size_t degBytes = ((size_t)N * sizeof(float) + 255) & ~(size_t)255;
  size_t nodeMat  = (size_t)N * HIDDEN * sizeof(float);
  float*  deg    = (float*)ws;
  float*  agg    = (float*)(ws + degBytes);
  float*  h      = (float*)(ws + degBytes + nodeMat);
  __bf16* WlBf   = (__bf16*)(ws + degBytes + 2 * nodeMat);
  __bf16* WrBf   = WlBf + (size_t)HIDDEN * IN_FEATS;
  __bf16* WoutBf = WrBf + (size_t)HIDDEN * IN_FEATS;
  float*  hout   = agg;

  hipMemsetAsync(deg, 0, (size_t)N * sizeof(float), stream);
  hipMemsetAsync(agg, 0, nodeMat, stream);
  hipMemsetAsync(out, 0, (size_t)N * NCLS * sizeof(float), stream);

  {  // stage weights as bf16 (tiny)
    int nWl = HIDDEN * IN_FEATS, nWo = NCLS * HIDDEN;
    cvt_bf16_kernel<<<(nWl + 255) / 256, 256, 0, stream>>>(Wl, WlBf, nWl);
    cvt_bf16_kernel<<<(nWl + 255) / 256, 256, 0, stream>>>(Wr, WrBf, nWl);
    cvt_bf16_kernel<<<(nWo + 255) / 256, 256, 0, stream>>>(Wout, WoutBf, nWo);
  }
  {
    int th = 256, bl = (E + th - 1) / th;
    degree_kernel<<<bl, th, 0, stream>>>(dst, deg, E);
  }
  {
    long long tot = (long long)E * 32;
    int th = 256;
    unsigned bl = (unsigned)((tot + th - 1) / th);
    scatter_feat_kernel<<<bl, th, 0, stream>>>(feat, src, dst, agg, E);
  }
  {
    long long waves = ((long long)(N + 15) / 16) * (HIDDEN / 16);
    int th = 256;
    unsigned bl = (unsigned)((waves * 32 + th - 1) / th);
    gemm1_kernel<<<bl, th, 0, stream>>>(agg, feat, deg, WlBf, WrBf,
                                        mask1, mask2, h, N);
  }
  {
    long long waves = ((long long)(N + 15) / 16) * (NCLS / 16);
    int th = 256;
    unsigned bl = (unsigned)((waves * 32 + th - 1) / th);
    gemm2_kernel<<<bl, th, 0, stream>>>(h, WoutBf, hout, N);
  }
  {
    long long tot = (long long)E * 32;
    int th = 256;
    unsigned bl = (unsigned)((tot + th - 1) / th);
    scatter_out_kernel<<<bl, th, 0, stream>>>(hout, src, dst, out, E);
  }
}
